// SequenceEncoder_65592740545103
// MI455X (gfx1250) — compile-verified, matrix operand order checked by reading, not run on hardware
//
#include <hip/hip_runtime.h>
#include <hip/hip_bf16.h>

// ---------------------------------------------------------------------------
// CDNA5 (gfx1250) Evoformer-style sequence encoder.
// All dense GEMMs via v_wmma_f32_16x16x32_f16 (f16 in, f32 accum).
// Rel-pos one-hot einsums computed analytically (lookup / prefix sums).
// Dims: B=16 N=256 CS=512 H=8 CH=64 K=32 BINS=65 CT=2048
// ---------------------------------------------------------------------------

typedef __attribute__((ext_vector_type(2)))  _Float16 v2h;
typedef __attribute__((ext_vector_type(4)))  _Float16 v4h;
typedef __attribute__((ext_vector_type(8)))  _Float16 v8h;
typedef __attribute__((ext_vector_type(16))) _Float16 v16h;
typedef __attribute__((ext_vector_type(8)))  float    v8f;

__device__ __forceinline__ v8f wmma32(v16h a, v16h b, v8f c) {
  return __builtin_amdgcn_wmma_f32_16x16x32_f16(
      /*neg_a=*/false, a, /*neg_b=*/false, b,
      /*c_mod=*/(short)0, c, /*reuse_a=*/false, /*reuse_b=*/false);
}

// One lane's 16-half fragment from an LDS row (halfs). Per ISA 16-bit A
// layout: lane group g owns runs [koff+8g,+8) -> VGPR0-3 and
// [koff+16+8g,+8) -> VGPR4-7.  (Two ds_load_b128.)
__device__ __forceinline__ v16h frag_h(const _Float16* rowptr, int koff, int g) {
  v8h lo = *(const v8h*)(rowptr + koff + 8 * g);
  v8h hi = *(const v8h*)(rowptr + koff + 16 + 8 * g);
  return __builtin_shufflevector(lo, hi, 0, 1, 2, 3, 4, 5, 6, 7,
                                 8, 9, 10, 11, 12, 13, 14, 15);
}

// Same, but source row is f32 in LDS: four b128 loads + converts.
__device__ __forceinline__ v16h frag_f(const float* rowptr, int koff, int g) {
  const float4 a0 = *(const float4*)(rowptr + koff + 8 * g);
  const float4 a1 = *(const float4*)(rowptr + koff + 8 * g + 4);
  const float4 b0 = *(const float4*)(rowptr + koff + 16 + 8 * g);
  const float4 b1 = *(const float4*)(rowptr + koff + 16 + 8 * g + 4);
  v16h f;
  f[0] = (_Float16)a0.x;  f[1] = (_Float16)a0.y;
  f[2] = (_Float16)a0.z;  f[3] = (_Float16)a0.w;
  f[4] = (_Float16)a1.x;  f[5] = (_Float16)a1.y;
  f[6] = (_Float16)a1.z;  f[7] = (_Float16)a1.w;
  f[8] = (_Float16)b0.x;  f[9] = (_Float16)b0.y;
  f[10] = (_Float16)b0.z; f[11] = (_Float16)b0.w;
  f[12] = (_Float16)b1.x; f[13] = (_Float16)b1.y;
  f[14] = (_Float16)b1.z; f[15] = (_Float16)b1.w;
  return f;
}

// ---------------------------------------------------------------------------
// Generic GEMM: C[M,N] = A[M,K] @ B[K,N] (+bias). f32 in/out, f16 WMMA.
// M,N multiples of 64; K arbitrary (guarded tail path).
// mode: 0 = plain, 1 = +relu, 2 = qkv head-permute store
//   (mode 2: col = d*8+h, row = b*256+i  ->  C[((b*8+h)*256+i)*64 + d])
// Block: 256 threads = 8 waves; 64x64 C tile; wave = 16x32 sub-tile.
// ---------------------------------------------------------------------------
#define LDSTR 40  // padded LDS row stride in halfs (80B; 16B-aligned runs)

__global__ __launch_bounds__(256) void gemm_kernel(
    const float* __restrict__ A, const float* __restrict__ Bw,
    float* __restrict__ C, int M, int K, int N,
    const float* __restrict__ bias, int mode) {
  __shared__ _Float16 As[64 * LDSTR];
  __shared__ _Float16 Bs[64 * LDSTR];  // column-major: Bs[n][k]
  const int tid = threadIdx.x;
  const int lane = tid & 31, wave = tid >> 5;
  const int g = lane >> 4, ln = lane & 15;
  const int rw = wave & 3, cw = wave >> 2;
  const int m0 = blockIdx.y * 64, n0 = blockIdx.x * 64;

  // fast-path staging indices (computed once)
  const int ra0 = tid >> 3, kq0 = (tid & 7) << 2;           // A chunk 0
  const int ra1 = (tid + 256) >> 3, kq1 = kq0;              // A chunk 1
  const int nq = (tid & 15) << 2, kb = (tid >> 4) << 1;     // B 2k x 4n chunk

  v8f acc0 = {}, acc1 = {};
  for (int k0 = 0; k0 < K; k0 += 32) {
    if (k0 + 32 <= K) {
      // ---- vectorized staging: issue all loads, then convert/store ----
      const float4 va0 = *(const float4*)&A[(size_t)(m0 + ra0) * K + k0 + kq0];
      const float4 va1 = *(const float4*)&A[(size_t)(m0 + ra1) * K + k0 + kq1];
      const float4 vb0 = *(const float4*)&Bw[(size_t)(k0 + kb) * N + n0 + nq];
      const float4 vb1 = *(const float4*)&Bw[(size_t)(k0 + kb + 1) * N + n0 + nq];
      v4h ha0 = {(_Float16)va0.x, (_Float16)va0.y, (_Float16)va0.z, (_Float16)va0.w};
      v4h ha1 = {(_Float16)va1.x, (_Float16)va1.y, (_Float16)va1.z, (_Float16)va1.w};
      *(v4h*)&As[ra0 * LDSTR + kq0] = ha0;
      *(v4h*)&As[ra1 * LDSTR + kq1] = ha1;
      v2h p0 = {(_Float16)vb0.x, (_Float16)vb1.x};
      v2h p1 = {(_Float16)vb0.y, (_Float16)vb1.y};
      v2h p2 = {(_Float16)vb0.z, (_Float16)vb1.z};
      v2h p3 = {(_Float16)vb0.w, (_Float16)vb1.w};
      *(v2h*)&Bs[(nq + 0) * LDSTR + kb] = p0;
      *(v2h*)&Bs[(nq + 1) * LDSTR + kb] = p1;
      *(v2h*)&Bs[(nq + 2) * LDSTR + kb] = p2;
      *(v2h*)&Bs[(nq + 3) * LDSTR + kb] = p3;
    } else {
      // ---- tail (only K=1032 last step): guarded scalar staging ----
      for (int idx = tid; idx < 2048; idx += 256) {
        int r = idx >> 5, ka = idx & 31;
        float va = (k0 + ka < K) ? A[(size_t)(m0 + r) * K + (k0 + ka)] : 0.f;
        As[r * LDSTR + ka] = (_Float16)va;
        int kk = idx >> 6, nb = idx & 63;
        float vb = (k0 + kk < K) ? Bw[(size_t)(k0 + kk) * N + (n0 + nb)] : 0.f;
        Bs[nb * LDSTR + kk] = (_Float16)vb;
      }
    }
    __syncthreads();
    v16h af = frag_h(&As[(rw * 16 + ln) * LDSTR], 0, g);
    v16h bf0 = frag_h(&Bs[(cw * 32 + ln) * LDSTR], 0, g);
    v16h bf1 = frag_h(&Bs[(cw * 32 + 16 + ln) * LDSTR], 0, g);
    acc0 = wmma32(af, bf0, acc0);
    acc1 = wmma32(af, bf1, acc1);
    __syncthreads();
  }
#pragma unroll
  for (int r = 0; r < 8; r++) {
    int row = m0 + rw * 16 + r + 8 * g;
    int c0 = n0 + cw * 32 + ln;
    int c1 = c0 + 16;
    float v0 = acc0[r], v1 = acc1[r];
    if (bias) { v0 += bias[c0]; v1 += bias[c1]; }
    if (mode == 1) { v0 = fmaxf(v0, 0.f); v1 = fmaxf(v1, 0.f); }
    if (mode == 2) {
      int bb = row >> 8, ii = row & 255;
      C[(size_t)(((bb << 3) | (c0 & 7)) * 256 + ii) * 64 + (c0 >> 3)] = v0;
      C[(size_t)(((bb << 3) | (c1 & 7)) * 256 + ii) * 64 + (c1 >> 3)] = v1;
    } else {
      C[(size_t)row * N + c0] = v0;
      C[(size_t)row * N + c1] = v1;
    }
  }
}

// ---------------------------------------------------------------------------
// Fused attention: one wave per (b, h, 16-row i-tile).
// q/k/v are head-permuted: x[(b*8+h)*256 + tok][64] (contiguous d).
// ---------------------------------------------------------------------------
__global__ __launch_bounds__(32) void attn_kernel(
    const float* __restrict__ qg, const float* __restrict__ kg,
    const float* __restrict__ vg, const unsigned char* __restrict__ mask,
    const float* __restrict__ Wb, float* __restrict__ cat,
    float* __restrict__ a_out) {
  __shared__ _Float16 qs[16 * 64];   // [i][d]
  __shared__ _Float16 ks[16 * 64];   // [j][d]
  __shared__ _Float16 vt[64 * 256];  // [d][j]
  __shared__ float ab[16 * 256];     // logits -> probs

  const int it = blockIdx.x, h = blockIdx.y, b = blockIdx.z;
  const int i0 = it * 16;
  const int lane = threadIdx.x, g = lane >> 4, ln = lane & 15;
  const size_t hb = (size_t)((b * 8 + h) * 256) * 64;
  const float* qp = qg + hb;
  const float* kp = kg + hb;
  const float* vp = vg + hb;

  // q tile: 256 float4 chunks
#pragma unroll
  for (int t = 0; t < 8; t++) {
    int idx4 = lane + t * 32;
    int i = idx4 >> 4, dq = (idx4 & 15) << 2;
    float4 x = *(const float4*)&qp[(size_t)(i0 + i) * 64 + dq];
    v4h hx = {(_Float16)x.x, (_Float16)x.y, (_Float16)x.z, (_Float16)x.w};
    *(v4h*)&qs[i * 64 + dq] = hx;
  }
  // v transposed: chunks of (2j x 4d) -> v2h dword stores
  for (int t = 0; t < 64; t++) {
    int c = lane + t * 32;
    int j2 = (c & 127) << 1, dq = (c >> 7) << 2;
    float4 x0 = *(const float4*)&vp[(size_t)j2 * 64 + dq];
    float4 x1 = *(const float4*)&vp[(size_t)(j2 + 1) * 64 + dq];
    v2h p0 = {(_Float16)x0.x, (_Float16)x1.x};
    v2h p1 = {(_Float16)x0.y, (_Float16)x1.y};
    v2h p2 = {(_Float16)x0.z, (_Float16)x1.z};
    v2h p3 = {(_Float16)x0.w, (_Float16)x1.w};
    *(v2h*)&vt[(dq + 0) * 256 + j2] = p0;
    *(v2h*)&vt[(dq + 1) * 256 + j2] = p1;
    *(v2h*)&vt[(dq + 2) * 256 + j2] = p2;
    *(v2h*)&vt[(dq + 3) * 256 + j2] = p3;
  }
  __syncthreads();

  // ---- logits = W_L*(q.k/8 + Wb[bin(j-i),h]*m) - 1e9*(1-m) ----
  for (int jt = 0; jt < 16; jt++) {
#pragma unroll
    for (int t = 0; t < 8; t++) {
      int idx4 = lane + t * 32;
      int j = idx4 >> 4, dq = (idx4 & 15) << 2;
      float4 x = *(const float4*)&kp[(size_t)(jt * 16 + j) * 64 + dq];
      v4h hx = {(_Float16)x.x, (_Float16)x.y, (_Float16)x.z, (_Float16)x.w};
      *(v4h*)&ks[j * 64 + dq] = hx;
    }
    __syncthreads();
    v8f c = {};
    c = wmma32(frag_h(&qs[ln * 64], 0, g), frag_h(&ks[ln * 64], 0, g), c);
    c = wmma32(frag_h(&qs[ln * 64], 32, g), frag_h(&ks[ln * 64], 32, g), c);
#pragma unroll
    for (int r = 0; r < 8; r++) {
      int il = r + 8 * g;
      int i = i0 + il;
      int j = jt * 16 + ln;
      int dd = j - i;
      int bin = (dd < -32 ? -32 : (dd > 32 ? 32 : dd)) + 32;
      bool mp = mask[b * 256 + i] && mask[b * 256 + j];
      float bias = mp ? Wb[bin * 8 + h] : 0.f;
      float lg = 0.70710678118f * (c[r] * 0.125f + bias) - (mp ? 0.f : 1e9f);
      ab[il * 256 + j] = lg;
    }
    __syncthreads();
  }

  // ---- softmax over j: 2 lanes per row (128 cols each), shfl_xor(16) ----
  {
    const int row = ln, seg = g;
    float* rp = &ab[row * 256 + seg * 128];
    float mx = -3.4e38f;
    for (int j = 0; j < 128; j += 4) {
      float4 x = *(float4*)&rp[j];
      mx = fmaxf(mx, fmaxf(fmaxf(x.x, x.y), fmaxf(x.z, x.w)));
    }
    mx = fmaxf(mx, __shfl_xor(mx, 16));
    float sum = 0.f;
    for (int j = 0; j < 128; j += 4) {
      float4 x = *(float4*)&rp[j];
      x.x = __expf(x.x - mx); x.y = __expf(x.y - mx);
      x.z = __expf(x.z - mx); x.w = __expf(x.w - mx);
      *(float4*)&rp[j] = x;
      sum += x.x + x.y + x.z + x.w;
    }
    sum += __shfl_xor(sum, 16);
    float inv = 1.f / sum;
    float* ao = &a_out[(size_t)((b * 8 + h) * 256 + i0 + row) * 256 + seg * 128];
    for (int j = 0; j < 128; j += 4) {
      float4 v = *(float4*)&rp[j];
      v.x *= inv; v.y *= inv; v.z *= inv; v.w *= inv;
      *(float4*)&rp[j] = v;
      *(float4*)&ao[j] = v;
    }
  }
  __syncthreads();

  // ---- o_pair[i,c]: c=1..63 -> single j=i+c-32; c=0 prefix; c=64 suffix ----
  {
    const int row = ln, i = i0 + ln;
    const float mi = mask[b * 256 + i] ? 1.f : 0.f;
    float* catrow = &cat[(size_t)(b * 256 + i) * 1032 + h * 65];
    int cbeg = g ? 32 : 0, cend = g ? 65 : 32;
    for (int cc = cbeg; cc < cend; cc++) {
      float v = 0.f;
      if (cc == 0) {
        for (int j = 0; j + 32 <= i; j++)
          v += ab[row * 256 + j] * (mask[b * 256 + j] ? mi : 0.f);
      } else if (cc == 64) {
        for (int j = i + 32; j < 256; j++)
          v += ab[row * 256 + j] * (mask[b * 256 + j] ? mi : 0.f);
      } else {
        int j = i + cc - 32;
        if (j >= 0 && j < 256)
          v = ab[row * 256 + j] * (mask[b * 256 + j] ? mi : 0.f);
      }
      catrow[cc] = v;
    }
  }
  __syncthreads();

  // ---- o = a @ v ----
  for (int dt = 0; dt < 4; dt++) {
    v8f oc = {};
    const _Float16* vrow = &vt[(dt * 16 + ln) * 256];
    const float* arow = &ab[ln * 256];
#pragma unroll
    for (int jt = 0; jt < 8; jt++)
      oc = wmma32(frag_f(arow, jt * 32, g), frag_h(vrow, jt * 32, g), oc);
#pragma unroll
    for (int r = 0; r < 8; r++) {
      int il = r + 8 * g;
      cat[(size_t)(b * 256 + i0 + il) * 1032 + 520 + h * 64 + dt * 16 + ln] =
          oc[r];
    }
  }
}

// ---------------------------------------------------------------------------
// out = LayerNorm(x + y) * gamma + beta ; one wave per token (512 features).
// ---------------------------------------------------------------------------
__global__ __launch_bounds__(32) void add_ln_kernel(
    const float* __restrict__ x, const float* __restrict__ y,
    const float* __restrict__ gamma, const float* __restrict__ beta,
    float* __restrict__ out) {
  const int row = blockIdx.x;
  const int lane = threadIdx.x;
  float v[16];
  float s = 0.f;
#pragma unroll
  for (int t = 0; t < 16; t++) {
    int e = lane + t * 32;
    v[t] = x[(size_t)row * 512 + e] + y[(size_t)row * 512 + e];
    s += v[t];
  }
#pragma unroll
  for (int o = 16; o > 0; o >>= 1) s += __shfl_xor(s, o);
  float mu = s * (1.f / 512.f);
  float q = 0.f;
#pragma unroll
  for (int t = 0; t < 16; t++) {
    float d = v[t] - mu;
    q += d * d;
  }
#pragma unroll
  for (int o = 16; o > 0; o >>= 1) q += __shfl_xor(q, o);
  float rs = rsqrtf(q * (1.f / 512.f) + 1e-5f);
#pragma unroll
  for (int t = 0; t < 16; t++) {
    int e = lane + t * 32;
    out[(size_t)row * 512 + e] = (v[t] - mu) * rs * gamma[e] + beta[e];
  }
}

// ---------------------------------------------------------------------------
extern "C" void kernel_launch(void* const* d_in, const int* in_sizes, int n_in,
                              void* d_out, int out_size, void* d_ws,
                              size_t ws_size, hipStream_t stream) {
  (void)in_sizes; (void)n_in; (void)out_size; (void)ws_size;
  const float* s = (const float*)d_in[0];
  const unsigned char* mask = (const unsigned char*)d_in[1];
  const float* Wq = (const float*)d_in[2];
  const float* Wk = (const float*)d_in[3];
  const float* Wv = (const float*)d_in[4];
  const float* Wb = (const float*)d_in[5];
  const float* Wo = (const float*)d_in[6];
  const float* bo = (const float*)d_in[7];
  const float* g1 = (const float*)d_in[8];
  const float* be1 = (const float*)d_in[9];
  const float* W1 = (const float*)d_in[10];
  const float* bf1 = (const float*)d_in[11];
  const float* W2 = (const float*)d_in[12];
  const float* bf2 = (const float*)d_in[13];
  const float* g2 = (const float*)d_in[14];
  const float* be2 = (const float*)d_in[15];

  // Workspace (floats): q|k|v (2M each, head-permuted) | cat | ffh
  float* ws = (float*)d_ws;
  float* q = ws;                 // 2,097,152
  float* kk = ws + 2097152;      // 2,097,152
  float* vv = ws + 4194304;      // 2,097,152
  float* cat = ws + 6291456;     // 4,227,072 (4096x1032)
  float* ffh = ws + 10518528;    // 8,388,608 (4096x2048)
  float* embd = q;               // q dead after attention
  float* s1 = kk;                // k dead after attention
  float* ff = vv;                // v dead after attention
  float* s2 = (float*)d_out;
  float* a_out = (float*)d_out + 2097152;

  dim3 blk(256);
  // QKV projections with head-permuted stores
  gemm_kernel<<<dim3(8, 64), blk, 0, stream>>>(s, Wq, q, 4096, 512, 512, nullptr, 2);
  gemm_kernel<<<dim3(8, 64), blk, 0, stream>>>(s, Wk, kk, 4096, 512, 512, nullptr, 2);
  gemm_kernel<<<dim3(8, 64), blk, 0, stream>>>(s, Wv, vv, 4096, 512, 512, nullptr, 2);
  // Fused attention -> a_out + cat[4096,1032]
  attn_kernel<<<dim3(16, 8, 16), dim3(32), 0, stream>>>(q, kk, vv, mask, Wb, cat, a_out);
  // Output projection: [4096,1032] @ [1032,512] + bo
  gemm_kernel<<<dim3(8, 64), blk, 0, stream>>>(cat, Wo, embd, 4096, 1032, 512, bo, 0);
  // s1 = LN(s + embd)
  add_ln_kernel<<<dim3(4096), dim3(32), 0, stream>>>(s, embd, g1, be1, s1);
  // FFN: relu(s1@W1+bf1) @ W2 + bf2
  gemm_kernel<<<dim3(32, 64), blk, 0, stream>>>(s1, W1, ffh, 4096, 512, 2048, bf1, 1);
  gemm_kernel<<<dim3(8, 64), blk, 0, stream>>>(ffh, W2, ff, 4096, 2048, 512, bf2, 0);
  // s2 = LN(s1 + ff)
  add_ln_kernel<<<dim3(4096), dim3(32), 0, stream>>>(s1, ff, g2, be2, s2);
}
